// MemoryAugmentedNetwork_88210038325293
// MI455X (gfx1250) — matrix operand order repaired and checked by reading, not compiled, not run on hardware
//
#include <hip/hip_runtime.h>
#include <math.h>

// ---------------------------------------------------------------------------
// MemoryAugmentedNetwork on MI455X (gfx1250, wave32, WMMA bf16 path)
//   query:         [B=131072, 256] f32
//   memory_keys:   [512, 256]      f32
//   memory_values: [512, 256]      f32
// outputs (concat in d_out, f32):
//   output  [B,256], weights [B,512], same_scene [B-1]
// ---------------------------------------------------------------------------

typedef __attribute__((ext_vector_type(16))) __bf16 v16bf;
typedef __attribute__((ext_vector_type(8)))  float  v8f;

#define MEM_M 512
#define MEM_D 256
#define ROWS_PER_BLOCK 128
#define SOFTMAX_TEMP 10.0f
#define NORM_EPS 1e-12f
#define SCENE_TH 0.8f

// float -> bf16, round-to-nearest-even
static __device__ __forceinline__ __bf16 f2bf(float f) {
  unsigned u = __builtin_bit_cast(unsigned, f);
  unsigned r = (u + 0x7FFFu + ((u >> 16) & 1u)) >> 16;
  return __builtin_bit_cast(__bf16, (unsigned short)r);
}

// load two 8-bf16 runs p[0..7] and p[16..23] (16B each, 16B-aligned) as a WMMA operand vector
struct B32Bytes { int4 a, b; };
static __device__ __forceinline__ v16bf ldb16(const __bf16* p) {
  B32Bytes t;
  t.a = *(const int4*)(p);
  t.b = *(const int4*)(p + 16);
  return __builtin_bit_cast(v16bf, t);
}

// pack 16 scaled floats (4x float4) into bf16 WMMA A-operand vector
static __device__ __forceinline__ v16bf pack16(float4 a, float4 b, float4 c,
                                               float4 d, float s) {
  v16bf r;
  r[0]  = f2bf(a.x * s); r[1]  = f2bf(a.y * s); r[2]  = f2bf(a.z * s); r[3]  = f2bf(a.w * s);
  r[4]  = f2bf(b.x * s); r[5]  = f2bf(b.y * s); r[6]  = f2bf(b.z * s); r[7]  = f2bf(b.w * s);
  r[8]  = f2bf(c.x * s); r[9]  = f2bf(c.y * s); r[10] = f2bf(c.z * s); r[11] = f2bf(c.w * s);
  r[12] = f2bf(d.x * s); r[13] = f2bf(d.y * s); r[14] = f2bf(d.z * s); r[15] = f2bf(d.w * s);
  return r;
}

// ---------------------------------------------------------------------------
// Kernel 0: normalize memory_keys -> bf16 Kn [512][256],
//           transpose memory_values -> bf16 Vt [256][512]
// one block per memory slot, 256 threads
// ---------------------------------------------------------------------------
__global__ void man_prep(const float* __restrict__ keys,
                         const float* __restrict__ vals,
                         __bf16* __restrict__ knorm,   // [512][256]
                         __bf16* __restrict__ vt) {    // [256][512]
  const int m = blockIdx.x;   // memory index 0..511
  const int t = threadIdx.x;  // dim index   0..255
  const float k = keys[m * MEM_D + t];

  float p = k * k;
  #pragma unroll
  for (int s = 1; s < 32; s <<= 1) p += __shfl_xor(p, s, 32);

  __shared__ float ws8[8];
  const int wave = t >> 5, lane = t & 31;
  if (lane == 0) ws8[wave] = p;
  __syncthreads();
  if (t == 0) {
    float s = 0.f;
    #pragma unroll
    for (int i = 0; i < 8; ++i) s += ws8[i];
    ws8[0] = s;
  }
  __syncthreads();
  const float inv = 1.0f / fmaxf(sqrtf(ws8[0]), NORM_EPS);

  knorm[m * MEM_D + t] = f2bf(k * inv);
  vt[t * MEM_M + m]    = f2bf(vals[m * MEM_D + t]);
}

// ---------------------------------------------------------------------------
// Kernel 1: fused normalize(query) -> GEMM1 (WMMA bf16) -> softmax (LDS)
//           -> weights out -> GEMM2 (WMMA bf16) -> output
// block = 256 threads = 8 waves; wave owns 16 rows; block owns 128 rows.
// dynamic LDS: sim[128][512] f32 (+ invsum[128]) = ~257 KB (one WG / WGP)
// ---------------------------------------------------------------------------
__global__ void __launch_bounds__(256)
man_main(const float* __restrict__ query,
         const __bf16* __restrict__ knorm,   // [512][256]
         const __bf16* __restrict__ vt,      // [256][512]
         float* __restrict__ out,            // [B][256]
         float* __restrict__ weights) {      // [B][512]
  extern __shared__ float smem[];
  float* sim    = smem;                  // [128][512]
  float* invsum = smem + ROWS_PER_BLOCK * MEM_M;  // [128]

  const int wave = threadIdx.x >> 5;
  const int lane = threadIdx.x & 31;
  const int lrow = lane & 15;        // row inside the wave's 16-row tile
  const bool hi  = lane >= 16;       // upper half-wave (A K-offset +8, B K-offset +8)
  const int r0   = blockIdx.x * ROWS_PER_BLOCK + wave * 16;
  const int grow = r0 + lrow;        // this lane's global query row

  const float* qrow = query + (size_t)grow * MEM_D;

  // ---- pass 1: row L2 norm (lanes L and L+16 split the 256 columns) ----
  float ss = 0.f;
  {
    const int cb = hi ? 128 : 0;
    #pragma unroll
    for (int i = 0; i < 128; i += 4) {
      float4 v = *(const float4*)(qrow + cb + i);
      ss += v.x * v.x + v.y * v.y + v.z * v.z + v.w * v.w;
    }
  }
  ss += __shfl_xor(ss, 16, 32);
  const float qinv = 1.0f / fmaxf(sqrtf(ss), NORM_EPS);

  // ---- pass 2: build 8 A-tiles (16x32 bf16) per ISA A-operand layout ----
  v16bf A1[8];
  {
    const int off = hi ? 8 : 0;
    #pragma unroll
    for (int t = 0; t < 8; ++t) {
      float4 u0 = *(const float4*)(qrow + 32 * t + off);
      float4 u1 = *(const float4*)(qrow + 32 * t + off + 4);
      float4 u2 = *(const float4*)(qrow + 32 * t + 16 + off);
      float4 u3 = *(const float4*)(qrow + 32 * t + 16 + off + 4);
      A1[t] = pack16(u0, u1, u2, u3, qinv);
    }
  }

  // ---- GEMM1: sim[16 rows][512] = Qn @ Kn^T via v_wmma_f32_16x16x32_bf16 ----
  const int ldsbase = wave * 16 * MEM_M;
  {
    const int khalf = hi ? 8 : 0;
    for (int nt = 0; nt < 32; ++nt) {
      const __bf16* bptr = knorm + (size_t)(nt * 16 + lrow) * MEM_D + khalf;
      v8f acc = {};
      #pragma unroll
      for (int kt = 0; kt < 8; ++kt) {
        v16bf b = ldb16(bptr + kt * 32);
        acc = __builtin_amdgcn_wmma_f32_16x16x32_bf16(
            false, A1[kt], false, b, (short)0, acc, false, false);
      }
      const int colb = nt * 16 + lrow;
      const int rb   = ldsbase + (hi ? 8 : 0) * MEM_M;
      #pragma unroll
      for (int j = 0; j < 8; ++j) sim[rb + j * MEM_M + colb] = acc[j];
    }
  }

  // ---- softmax over each of the wave's 16 rows (all in this wave's LDS) ----
  float wInv;  // 1/rowsum for this lane's row (lrow)
  {
    const int myrow = ldsbase + lrow * MEM_M + (hi ? 256 : 0);
    float m = -3.0e38f;
    #pragma unroll 8
    for (int i = 0; i < 256; i += 4) {
      float4 v = *(const float4*)&sim[myrow + i];
      m = fmaxf(m, fmaxf(fmaxf(v.x, v.y), fmaxf(v.z, v.w)));
    }
    m = fmaxf(m, __shfl_xor(m, 16, 32));

    const float c = SOFTMAX_TEMP * 1.4426950408889634f;  // temp * log2(e)
    float s = 0.f;
    #pragma unroll 8
    for (int i = 0; i < 256; i += 4) {
      float4 v = *(const float4*)&sim[myrow + i];
      float4 e;
      e.x = exp2f((v.x - m) * c);
      e.y = exp2f((v.y - m) * c);
      e.z = exp2f((v.z - m) * c);
      e.w = exp2f((v.w - m) * c);
      s += e.x + e.y + e.z + e.w;
      *(float4*)&sim[myrow + i] = e;  // keep exp values for GEMM2
    }
    s += __shfl_xor(s, 16, 32);
    wInv = 1.0f / s;
    if (!hi) invsum[wave * 16 + lrow] = wInv;
  }

  // ---- write normalized weights to global, fully coalesced b128 stores ----
  {
    float* wout = weights + (size_t)r0 * MEM_M;
    const float* lsrc = sim + ldsbase;
    for (int it = 0; it < 64; ++it) {
      const int flat = it * 128 + lane * 4;   // 0..8191 over 16x512
      const int row  = flat >> 9;
      const float is = invsum[wave * 16 + row];
      float4 e = *(const float4*)(lsrc + flat);
      float4 o = {e.x * is, e.y * is, e.z * is, e.w * is};
      *(float4*)(wout + flat) = o;
    }
  }

  // ---- GEMM2: out[16][256] = softmax @ V ;  A from LDS exp * wInv ----
  v16bf A2[16];
  {
    const int off = hi ? 8 : 0;
    const float* lrp = sim + ldsbase + lrow * MEM_M;
    #pragma unroll
    for (int t = 0; t < 16; ++t) {
      float4 u0 = *(const float4*)(lrp + 32 * t + off);
      float4 u1 = *(const float4*)(lrp + 32 * t + off + 4);
      float4 u2 = *(const float4*)(lrp + 32 * t + 16 + off);
      float4 u3 = *(const float4*)(lrp + 32 * t + 16 + off + 4);
      A2[t] = pack16(u0, u1, u2, u3, wInv);
    }
  }
  {
    const int khalf = hi ? 8 : 0;
    for (int nt = 0; nt < 16; ++nt) {
      const __bf16* bptr = vt + (size_t)(nt * 16 + lrow) * MEM_M + khalf;
      v8f acc = {};
      #pragma unroll
      for (int kt = 0; kt < 16; ++kt) {
        v16bf b = ldb16(bptr + kt * 32);
        acc = __builtin_amdgcn_wmma_f32_16x16x32_bf16(
            false, A2[kt], false, b, (short)0, acc, false, false);
      }
      const int colg = nt * 16 + lrow;
      const int rg   = r0 + (hi ? 8 : 0);
      #pragma unroll
      for (int j = 0; j < 8; ++j)
        out[(size_t)(rg + j) * MEM_D + colg] = acc[j];
    }
  }
}

// ---------------------------------------------------------------------------
// Kernel 2: scene change flags: ||q[i+1]-q[i]|| < 0.8  (one wave per pair)
// ---------------------------------------------------------------------------
__global__ void man_scene(const float* __restrict__ q,
                          float* __restrict__ flags, int nDiff) {
  const int wi   = blockIdx.x * 8 + (threadIdx.x >> 5);
  const int lane = threadIdx.x & 31;
  if (wi >= nDiff) return;
  const float* a = q + (size_t)wi * MEM_D + lane * 8;
  const float* b = a + MEM_D;
  float s = 0.f;
  #pragma unroll
  for (int i = 0; i < 8; i += 4) {
    float4 x = *(const float4*)(a + i);
    float4 y = *(const float4*)(b + i);
    float dx = y.x - x.x, dy = y.y - x.y, dz = y.z - x.z, dw = y.w - x.w;
    s += dx * dx + dy * dy + dz * dz + dw * dw;
  }
  #pragma unroll
  for (int m = 1; m < 32; m <<= 1) s += __shfl_xor(s, m, 32);
  if (lane == 0) flags[wi] = (sqrtf(s) < SCENE_TH) ? 1.0f : 0.0f;
}

// ---------------------------------------------------------------------------
extern "C" void kernel_launch(void* const* d_in, const int* in_sizes, int n_in,
                              void* d_out, int out_size, void* d_ws,
                              size_t ws_size, hipStream_t stream) {
  (void)n_in; (void)out_size; (void)ws_size;
  const float* query = (const float*)d_in[0];
  const float* keys  = (const float*)d_in[1];
  const float* vals  = (const float*)d_in[2];
  const int B = in_sizes[0] / MEM_D;  // 131072

  float* out     = (float*)d_out;
  float* weights = out + (size_t)B * MEM_D;
  float* scene   = weights + (size_t)B * MEM_M;

  __bf16* knorm = (__bf16*)d_ws;                       // 512*256*2 = 256 KB
  __bf16* vt    = knorm + (size_t)MEM_M * MEM_D;       // 256*512*2 = 256 KB

  man_prep<<<MEM_M, MEM_D, 0, stream>>>(keys, vals, knorm, vt);

  const size_t smemBytes =
      (size_t)(ROWS_PER_BLOCK * MEM_M + ROWS_PER_BLOCK) * sizeof(float);
  man_main<<<B / ROWS_PER_BLOCK, 256, smemBytes, stream>>>(query, knorm, vt,
                                                           out, weights);

  const int nDiff = B - 1;
  man_scene<<<(nDiff + 7) / 8, 256, 0, stream>>>(query, scene, nDiff);
}
